// ParallelFLAAttention_60060822667664
// MI455X (gfx1250) — matrix operand order repaired and checked by reading, not compiled, not run on hardware
//
#include <hip/hip_runtime.h>
#include <hip/hip_bf16.h>

typedef __attribute__((ext_vector_type(16))) __bf16 v16bf;
typedef __attribute__((ext_vector_type(8)))  float  v8f;

#define B_    2
#define SEQ_  1024
#define HID_  2048
#define NH_   32
#define NKV_  8
#define HD_   64
#define SCALE_ 0.125f
#define NROWS (B_*SEQ_)          // 2048
#define NQK   (HID_ + 2*NKV_*HD_) // 3072 (q | k | v columns)
#define KOFF  HID_               // 2048
#define VOFF  (HID_ + NKV_*HD_)  // 2560

__device__ __forceinline__ v8f wmma_bf16(v16bf a, v16bf b, v8f c) {
  return __builtin_amdgcn_wmma_f32_16x16x32_bf16(false, a, false, b, (short)0, c, false, false);
}

union U32B { uint4 u[2]; v16bf v; };

// ---------------------------------------------------------------- pack / convert
__global__ __launch_bounds__(256) void k_conv_h(const float* __restrict__ H, __bf16* __restrict__ Hbf, int n) {
  int i = blockIdx.x * 256 + threadIdx.x;
  if (i < n) Hbf[i] = (__bf16)H[i];
}

// WT[n][k] = concat(Wq,Wk,Wv)[k][n]  (transposed, N-major, K contiguous)
__global__ __launch_bounds__(256) void k_pack_qkv(const float* __restrict__ Wq, const float* __restrict__ Wk,
                                                  const float* __restrict__ Wv, __bf16* __restrict__ WT) {
  int i = blockIdx.x * 256 + threadIdx.x;
  if (i >= NQK * HID_) return;
  int nn = i / HID_, kk = i % HID_;
  float v;
  if (nn < KOFF)       v = Wq[(size_t)kk * HID_ + nn];
  else if (nn < VOFF)  v = Wk[(size_t)kk * (NKV_*HD_) + (nn - KOFF)];
  else                 v = Wv[(size_t)kk * (NKV_*HD_) + (nn - VOFF)];
  WT[(size_t)nn * HID_ + kk] = (__bf16)v;
}

__global__ __launch_bounds__(256) void k_pack_wo(const float* __restrict__ Wo, __bf16* __restrict__ WoT) {
  int i = blockIdx.x * 256 + threadIdx.x;
  if (i >= HID_ * HID_) return;
  int nn = i / HID_, kk = i % HID_;
  WoT[(size_t)nn * HID_ + kk] = (__bf16)Wo[(size_t)kk * HID_ + nn];
}

// ---------------------------------------------------------------- generic bf16 WMMA GEMM
// A[M,K] row-major bf16, BT[N,K] row-major bf16 (i.e. B transposed), C[M,N] f32.
// If fla != nullptr: C = alpha*acc + beta*fla (fused final epilogue).
__global__ __launch_bounds__(256) void k_gemm_bf16(const __bf16* __restrict__ A, const __bf16* __restrict__ BT,
                                                   float* __restrict__ C, const float* __restrict__ fla,
                                                   int M, int N, int K, float alpha, float beta) {
  int lane = threadIdx.x & 31;
  int wave = threadIdx.x >> 5;
  int tile = blockIdx.x * 8 + wave;
  int tpn  = N >> 4;
  int tm   = (tile / tpn) << 4;
  int tn   = (tile % tpn) << 4;
  if (tm >= M) return;
  int rc    = lane & 15;                  // A: row in tile, B: col in tile, C: col
  int abase = (lane < 16) ? 0 : 8;        // A k-offset pattern + C row base
  int bbase = (lane < 16) ? 0 : 8;        // B k-offset
  const __bf16* ap = A  + (size_t)(tm + rc) * K + abase;
  const __bf16* bp = BT + (size_t)(tn + rc) * K + bbase;
  v8f acc = {};
  for (int kc = 0; kc < K; kc += 32) {
    U32B ua, ub;
    ua.u[0] = *(const uint4*)(ap + kc);        // halves j<8  -> k = abase + j
    ua.u[1] = *(const uint4*)(ap + kc + 16);   // halves j>=8 -> k = abase + 16 + (j-8)
    ub.u[0] = *(const uint4*)(bp + kc);        // halves j<8  -> k = bbase + j
    ub.u[1] = *(const uint4*)(bp + kc + 16);   // halves j>=8 -> k = bbase + 16 + (j-8)
    acc = wmma_bf16(ua.v, ub.v, acc);
  }
  for (int r = 0; r < 8; ++r) {
    size_t idx = (size_t)(tm + abase + r) * N + tn + rc;
    float val = acc[r];
    if (fla) val = alpha * val + beta * fla[idx];
    C[idx] = val;
  }
}

// ---------------------------------------------------------------- preprocessing
// per (row, q-head): softmax(q) -> qf; raw q -> Qbf [b,h,s,d]
__global__ __launch_bounds__(256) void k_prep_q(const float* __restrict__ QKV, float* __restrict__ qf,
                                                __bf16* __restrict__ Qbf) {
  int idx = blockIdx.x * 256 + threadIdx.x;
  if (idx >= NROWS * NH_) return;
  int row = idx >> 5, h = idx & 31;
  int b = row >> 10, s = row & 1023;
  const float* qp = QKV + (size_t)row * NQK + h * HD_;
  float mx = -1e30f;
  for (int d = 0; d < HD_; ++d) mx = fmaxf(mx, qp[d]);
  float sum = 0.f;
  for (int d = 0; d < HD_; ++d) sum += __expf(qp[d] - mx);
  float inv = 1.0f / sum;
  float*  qo = qf + (size_t)row * HID_ + h * HD_;
  __bf16* qb = Qbf + ((size_t)(b * NH_ + h) * SEQ_ + s) * HD_;
  for (int d = 0; d < HD_; ++d) {
    qo[d] = __expf(qp[d] - mx) * inv;
    qb[d] = (__bf16)qp[d];
  }
}

// per (row, kv-head): softmax(k) -> kf; logsigmoid(k)/16 -> gf; raw k -> Kbf [b,kh,s,d]; v -> Vt [b,kh,d,s]
__global__ __launch_bounds__(256) void k_prep_kv(const float* __restrict__ QKV, float* __restrict__ kf,
                                                 float* __restrict__ gf, __bf16* __restrict__ Kbf,
                                                 __bf16* __restrict__ Vt) {
  int idx = blockIdx.x * 256 + threadIdx.x;
  if (idx >= NROWS * NKV_) return;
  int row = idx >> 3, kh = idx & 7;
  int b = row >> 10, s = row & 1023;
  const float* kp = QKV + (size_t)row * NQK + KOFF + kh * HD_;
  const float* vp = QKV + (size_t)row * NQK + VOFF + kh * HD_;
  float mx = -1e30f;
  for (int d = 0; d < HD_; ++d) mx = fmaxf(mx, kp[d]);
  float sum = 0.f;
  for (int d = 0; d < HD_; ++d) sum += __expf(kp[d] - mx);
  float inv = 1.0f / sum;
  float*  ko = kf + (size_t)row * (NKV_*HD_) + kh * HD_;
  float*  go = gf + (size_t)row * (NKV_*HD_) + kh * HD_;
  __bf16* kb = Kbf + ((size_t)(b * NKV_ + kh) * SEQ_ + s) * HD_;
  __bf16* vt = Vt  + ((size_t)(b * NKV_ + kh) * HD_) * SEQ_ + s;
  for (int d = 0; d < HD_; ++d) {
    float x = kp[d];
    ko[d] = __expf(x - mx) * inv;
    go[d] = (fminf(x, 0.f) - log1pf(__expf(-fabsf(x)))) * (1.0f / 16.0f);
    kb[d] = (__bf16)x;
    vt[(size_t)d * SEQ_] = (__bf16)vp[d];
  }
}

// ---------------------------------------------------------------- gated linear attention scan
// one block per (b,h); S[64][64] held in registers (16 cells/thread), broadcast via LDS
__global__ __launch_bounds__(256) void k_fla_scan(const float* __restrict__ qf, const float* __restrict__ kf,
                                                  const float* __restrict__ gf, const float* __restrict__ QKV,
                                                  float* __restrict__ fla) {
  __shared__ float sq[HD_], sk[HD_], sv[HD_], sg[HD_];
  __shared__ float red[4][HD_];
  int tid = threadIdx.x;
  int bh = blockIdx.x;
  int b = bh >> 5, h = bh & 31, kvh = h >> 2;
  int v = tid & 63, dq = tid >> 6;
  float S[16];
  for (int i = 0; i < 16; ++i) S[i] = 0.f;
  for (int s = 0; s < SEQ_; ++s) {
    size_t row = (size_t)b * SEQ_ + s;
    if (tid < HD_) {
      sq[tid] = qf[row * HID_ + h * HD_ + tid];
      sk[tid] = kf[row * (NKV_*HD_) + kvh * HD_ + tid];
      sg[tid] = gf[row * (NKV_*HD_) + kvh * HD_ + tid];
      sv[tid] = QKV[row * NQK + VOFF + kvh * HD_ + tid];
    }
    __syncthreads();
    float vt = sv[v];
    float acc = 0.f;
    for (int i = 0; i < 16; ++i) {
      int d = dq * 16 + i;
      S[i] = __expf(sg[d]) * S[i] + sk[d] * vt;
      acc += sq[d] * S[i];
    }
    red[dq][v] = acc;
    __syncthreads();
    if (dq == 0)
      fla[row * HID_ + h * HD_ + v] = SCALE_ * (red[0][v] + red[1][v] + red[2][v] + red[3][v]);
  }
}

// ---------------------------------------------------------------- flash causal MHA (bf16 WMMA)
// grid (B*NH, SEQ/128); 8 waves/block, wave owns a 16-row q tile
__global__ __launch_bounds__(256) void k_flash_mha(const __bf16* __restrict__ Qbf, const __bf16* __restrict__ Kbf,
                                                   const __bf16* __restrict__ Vt, __bf16* __restrict__ baseA) {
  __shared__ __align__(32) __bf16 Plds[8][16 * 32];
  int lane = threadIdx.x & 31;
  int w    = threadIdx.x >> 5;
  int bh = blockIdx.x;
  int b = bh >> 5, h = bh & 31, kvh = h >> 2;
  int qbase = blockIdx.y * 128 + w * 16;
  int n     = lane & 15;
  int abase = (lane < 16) ? 0 : 8;
  int bbase = (lane < 16) ? 0 : 8;
  const __bf16* Qp = Qbf + ((size_t)(b * NH_  + h)   * SEQ_ + qbase) * HD_;
  const __bf16* Kp = Kbf + ((size_t)(b * NKV_ + kvh) * SEQ_) * HD_;
  const __bf16* Vp = Vt  + ((size_t)(b * NKV_ + kvh) * HD_) * SEQ_;
  __bf16* Pw = &Plds[w][0];

  U32B aq[2];
  for (int c = 0; c < 2; ++c) {
    const __bf16* ap = Qp + (size_t)n * HD_ + c * 32 + abase;
    aq[c].u[0] = *(const uint4*)ap;
    aq[c].u[1] = *(const uint4*)(ap + 16);
  }
  v8f o0 = {}, o1 = {}, o2 = {}, o3 = {};
  float m[8], l[8];
  for (int r = 0; r < 8; ++r) { m[r] = -1e30f; l[r] = 0.f; }

  for (int k0 = 0; k0 <= qbase + 15; k0 += 32) {
    v8f s0 = {}, s1 = {};
    for (int c = 0; c < 2; ++c) {
      U32B bk0, bk1;
      const __bf16* bp0 = Kp + (size_t)(k0 + n) * HD_ + c * 32 + bbase;
      const __bf16* bp1 = bp0 + 16 * HD_;
      bk0.u[0] = *(const uint4*)bp0; bk0.u[1] = *(const uint4*)(bp0 + 16);
      bk1.u[0] = *(const uint4*)bp1; bk1.u[1] = *(const uint4*)(bp1 + 16);
      s0 = wmma_bf16(aq[c].v, bk0.v, s0);
      s1 = wmma_bf16(aq[c].v, bk1.v, s1);
    }
    // online softmax, row stats live in [8] regs (rows abase+r), replicated over 16 lanes
    for (int r = 0; r < 8; ++r) {
      int rowg = qbase + abase + r;
      float v0 = s0[r] * SCALE_; if (k0 + n      > rowg) v0 = -1e30f;
      float v1 = s1[r] * SCALE_; if (k0 + 16 + n > rowg) v1 = -1e30f;
      float rm = fmaxf(v0, v1);
      rm = fmaxf(rm, __shfl_xor(rm, 1, 32));
      rm = fmaxf(rm, __shfl_xor(rm, 2, 32));
      rm = fmaxf(rm, __shfl_xor(rm, 4, 32));
      rm = fmaxf(rm, __shfl_xor(rm, 8, 32));
      float mn = fmaxf(m[r], rm);
      float cc = __expf(m[r] - mn);
      float p0 = __expf(v0 - mn), p1 = __expf(v1 - mn);
      float rs = p0 + p1;
      rs += __shfl_xor(rs, 1, 32);
      rs += __shfl_xor(rs, 2, 32);
      rs += __shfl_xor(rs, 4, 32);
      rs += __shfl_xor(rs, 8, 32);
      l[r] = l[r] * cc + rs; m[r] = mn;
      o0[r] *= cc; o1[r] *= cc; o2[r] *= cc; o3[r] *= cc;
      Pw[(abase + r) * 32 + n]      = (__bf16)p0;
      Pw[(abase + r) * 32 + 16 + n] = (__bf16)p1;
    }
    // P: LDS -> A-layout fragment (16x32, bf16)
    U32B pa;
    const __bf16* pp = Pw + n * 32 + abase;
    pa.u[0] = *(const uint4*)pp;
    pa.u[1] = *(const uint4*)(pp + 16);
    // O += P @ V   (V d-major: contiguous key runs)
    {
      U32B bv;
      const __bf16* vp0 = Vp + (size_t)(0 * 16 + n) * SEQ_ + k0 + bbase;
      bv.u[0] = *(const uint4*)vp0; bv.u[1] = *(const uint4*)(vp0 + 16);
      o0 = wmma_bf16(pa.v, bv.v, o0);
      const __bf16* vp1 = Vp + (size_t)(1 * 16 + n) * SEQ_ + k0 + bbase;
      bv.u[0] = *(const uint4*)vp1; bv.u[1] = *(const uint4*)(vp1 + 16);
      o1 = wmma_bf16(pa.v, bv.v, o1);
      const __bf16* vp2 = Vp + (size_t)(2 * 16 + n) * SEQ_ + k0 + bbase;
      bv.u[0] = *(const uint4*)vp2; bv.u[1] = *(const uint4*)(vp2 + 16);
      o2 = wmma_bf16(pa.v, bv.v, o2);
      const __bf16* vp3 = Vp + (size_t)(3 * 16 + n) * SEQ_ + k0 + bbase;
      bv.u[0] = *(const uint4*)vp3; bv.u[1] = *(const uint4*)(vp3 + 16);
      o3 = wmma_bf16(pa.v, bv.v, o3);
    }
  }
  for (int r = 0; r < 8; ++r) {
    float inv = 1.0f / l[r];
    int rowg = qbase + abase + r;
    size_t orow = ((size_t)(b * SEQ_ + rowg)) * HID_ + h * HD_;
    baseA[orow + 0  + n] = (__bf16)(o0[r] * inv);
    baseA[orow + 16 + n] = (__bf16)(o1[r] * inv);
    baseA[orow + 32 + n] = (__bf16)(o2[r] * inv);
    baseA[orow + 48 + n] = (__bf16)(o3[r] * inv);
  }
}

// ---------------------------------------------------------------- launcher
extern "C" void kernel_launch(void* const* d_in, const int* in_sizes, int n_in,
                              void* d_out, int out_size, void* d_ws, size_t ws_size,
                              hipStream_t stream) {
  const float* H  = (const float*)d_in[0];
  const float* Wq = (const float*)d_in[1];
  const float* Wk = (const float*)d_in[2];
  const float* Wv = (const float*)d_in[3];
  const float* Wo = (const float*)d_in[4];
  float* out = (float*)d_out;

  char* ws = (char*)d_ws;
  size_t off = 0;
  auto take = [&](size_t bytes) { void* p = ws + off; off = (off + bytes + 255) & ~(size_t)255; return p; };
  __bf16* Hbf   = (__bf16*)take((size_t)NROWS * HID_ * 2);
  __bf16* WTqkv = (__bf16*)take((size_t)NQK * HID_ * 2);
  float*  QKV   = (float*) take((size_t)NROWS * NQK * 4);
  float*  qf    = (float*) take((size_t)NROWS * HID_ * 4);
  float*  kf    = (float*) take((size_t)NROWS * NKV_ * HD_ * 4);
  float*  gf    = (float*) take((size_t)NROWS * NKV_ * HD_ * 4);
  __bf16* Qbf   = (__bf16*)take((size_t)NROWS * NH_ * HD_ * 2);
  __bf16* Kbf   = (__bf16*)take((size_t)NROWS * NKV_ * HD_ * 2);
  __bf16* Vt    = (__bf16*)take((size_t)NROWS * NKV_ * HD_ * 2);
  float*  fla   = (float*) take((size_t)NROWS * HID_ * 4);
  __bf16* baseA = (__bf16*)take((size_t)NROWS * HID_ * 2);
  __bf16* WoT   = (__bf16*)take((size_t)HID_ * HID_ * 2);
  (void)ws_size; (void)n_in; (void)in_sizes; (void)out_size;

  // pack / convert
  k_conv_h<<<(NROWS * HID_ + 255) / 256, 256, 0, stream>>>(H, Hbf, NROWS * HID_);
  k_pack_qkv<<<(NQK * HID_ + 255) / 256, 256, 0, stream>>>(Wq, Wk, Wv, WTqkv);
  k_pack_wo<<<(HID_ * HID_ + 255) / 256, 256, 0, stream>>>(Wo, WoT);

  // fused QKV projection GEMM: [2048x2048] x [2048x3072]
  {
    int tiles = (NROWS / 16) * (NQK / 16);
    k_gemm_bf16<<<tiles / 8, 256, 0, stream>>>(Hbf, WTqkv, QKV, nullptr,
                                               NROWS, NQK, HID_, 1.f, 0.f);
  }

  // head preprocessing
  k_prep_q<<<(NROWS * NH_ + 255) / 256, 256, 0, stream>>>(QKV, qf, Qbf);
  k_prep_kv<<<(NROWS * NKV_ + 255) / 256, 256, 0, stream>>>(QKV, kf, gf, Kbf, Vt);

  // gated linear attention scan (64 blocks, one per (b,h))
  k_fla_scan<<<B_ * NH_, 256, 0, stream>>>(qf, kf, gf, QKV, fla);

  // causal flash MHA
  {
    dim3 grid(B_ * NH_, SEQ_ / 128);
    k_flash_mha<<<grid, 256, 0, stream>>>(Qbf, Kbf, Vt, baseA);
  }

  // final: out = 0.99*(base @ Wo) + 0.01*fla
  {
    int tiles = (NROWS / 16) * (HID_ / 16);
    k_gemm_bf16<<<tiles / 8, 256, 0, stream>>>(baseA, WoT, out, fla,
                                               NROWS, HID_, HID_, 0.99f, 0.01f);
  }
}
